// EmbeddingAndSeqformer_82884278878870
// MI455X (gfx1250) — matrix operand order repaired and checked by reading, not compiled, not run on hardware
//
#include <hip/hip_runtime.h>
#include <hip/hip_bf16.h>

// ---------------- CDNA5 WMMA types ----------------
typedef __attribute__((ext_vector_type(16))) _Float16 v16h;
typedef __attribute__((ext_vector_type(8)))  _Float16 v8h;
typedef __attribute__((ext_vector_type(8)))  float    v8f;
typedef __attribute__((ext_vector_type(4)))  unsigned int u32x4;

union FragU { v16h h; u32x4 q[2]; };

__device__ __forceinline__ v8f wmma_f16(v16h a, v16h b, v8f c) {
  // D = A(16x32 f16) * B(32x16 f16) + C(16x16 f32)
  return __builtin_amdgcn_wmma_f32_16x16x32_f16(false, a, false, b, (short)0, c, false, false);
}

__device__ __forceinline__ float redmax16(float v) {
  v = fmaxf(v, __shfl_xor(v, 1, 32));
  v = fmaxf(v, __shfl_xor(v, 2, 32));
  v = fmaxf(v, __shfl_xor(v, 4, 32));
  v = fmaxf(v, __shfl_xor(v, 8, 32));
  return v;
}
__device__ __forceinline__ float redsum16(float v) {
  v += __shfl_xor(v, 1, 32);
  v += __shfl_xor(v, 2, 32);
  v += __shfl_xor(v, 4, 32);
  v += __shfl_xor(v, 8, 32);
  return v;
}
__device__ __forceinline__ float redsum32(float v) {
  v += __shfl_xor(v, 1, 32);
  v += __shfl_xor(v, 2, 32);
  v += __shfl_xor(v, 4, 32);
  v += __shfl_xor(v, 8, 32);
  v += __shfl_xor(v, 16, 32);
  return v;
}

#define LSEQ 256
#define CCH  128
#define NH   4
#define DH   32
#define NPIX (LSEQ*LSEQ)          // 65536
#define MASKVAL (-3.4028235e34f)

// ---------------------------------------------------------------
// Kernel 1: LayerNorm over C + pair bias  bias[h][i][j] = xn . w_pair[:,h]
// grid = NPIX blocks of 128 threads; wave32 shuffle reductions + 4-elem LDS.
// ---------------------------------------------------------------
__global__ __launch_bounds__(128) void ln_bias_kernel(
    const float* __restrict__ x, const float* __restrict__ g,
    const float* __restrict__ b, const float* __restrict__ wp,
    _Float16* __restrict__ xn, float* __restrict__ bias)
{
  __shared__ float part[4];
  __shared__ float sBroadcast;
  int p = blockIdx.x, c = threadIdx.x;
  int wave = c >> 5, lane = c & 31;
  float v = x[p * CCH + c];

  // mean
  float ps = redsum32(v);
  if (lane == 0) part[wave] = ps;
  __syncthreads();
  if (c == 0) sBroadcast = (part[0] + part[1] + part[2] + part[3]) * (1.0f / CCH);
  __syncthreads();
  float mean = sBroadcast;
  float dv = v - mean;

  // variance
  ps = redsum32(dv * dv);
  __syncthreads();
  if (lane == 0) part[wave] = ps;
  __syncthreads();
  if (c == 0) sBroadcast = (part[0] + part[1] + part[2] + part[3]) * (1.0f / CCH);
  __syncthreads();
  float var = sBroadcast;

  float xv = dv * rsqrtf(var + 1e-5f) * g[c] + b[c];
  xn[p * CCH + c] = (_Float16)xv;

  for (int h = 0; h < NH; ++h) {
    float d = redsum32(xv * wp[c * NH + h]);
    __syncthreads();
    if (lane == 0) part[wave] = d;
    __syncthreads();
    if (c == 0) bias[h * NPIX + p] = part[0] + part[1] + part[2] + part[3];
  }
}

// ---------------------------------------------------------------
// Kernel 2: q/k/v/gate projections.  xn(65536x128 f16) @ W(128x128)
// grid = 512 blocks * 256 threads; each wave owns 16 rows.
// w-loop fully unrolled so dst pointers stay address-space-inferred
// (global_store, not flat_store) and the w==3 branch resolves statically.
// ---------------------------------------------------------------
__global__ __launch_bounds__(256) void gemm_qkvg_kernel(
    const _Float16* __restrict__ xn,
    const float* __restrict__ wq, const float* __restrict__ wk,
    const float* __restrict__ wv, const float* __restrict__ wg,
    const float* __restrict__ bg,
    float* __restrict__ qraw, float* __restrict__ kraw,
    float* __restrict__ vraw, _Float16* __restrict__ gate)
{
  __shared__ _Float16 wlds[CCH * CCH];   // 32 KB, transposed [n][k] f16
  int tid = threadIdx.x, wave = tid >> 5, lane = tid & 31;
  int rlo = lane & 15, khalf = lane >> 4;
  int row0 = blockIdx.x * 128 + wave * 16;
  int s  = row0 >> 8;                    // constant within the 16-row tile
  int l0 = (row0 & 255) + 8 * khalf;     // this lane's first l

  FragU A[4];
  for (int kc = 0; kc < 4; ++kc) {
    const _Float16* ap = xn + (row0 + rlo) * CCH + kc * 32 + khalf * 8;
    A[kc].q[0] = *(const u32x4*)ap;
    A[kc].q[1] = *(const u32x4*)(ap + 16);
  }

  const float* const Ws[4] = { wq, wk, wv, wg };
  float* const Dst[3] = { qraw, kraw, vraw };
#pragma unroll
  for (int w = 0; w < 4; ++w) {
    __syncthreads();
    for (int idx = tid; idx < CCH * CCH; idx += 256) wlds[(idx & (CCH - 1)) * CCH + (idx >> 7)] = (_Float16)Ws[w][idx];
    __syncthreads();
    int koff = rlo + 16 * khalf;                 // unused; B lane holds column n = nt*16+rlo, k along elements
    for (int nt = 0; nt < 8; ++nt) {
      FragU B[4];
      for (int kc = 0; kc < 4; ++kc) {           // preload all B frags
        const _Float16* bp = wlds + (nt * 16 + rlo) * CCH + kc * 32 + khalf * 8;
        B[kc].q[0] = *(const u32x4*)bp;
        B[kc].q[1] = *(const u32x4*)(bp + 16);
      }
      v8f acc = {};
      for (int kc = 0; kc < 4; ++kc) acc = wmma_f16(A[kc].h, B[kc].h, acc);

      int e = nt * 16 + rlo;                     // output column
      if (w == 3) {                              // gate: sigmoid, rows stride CCH
        float bgv = bg[e];
        _Float16* gp = gate + (row0 + 8 * khalf) * CCH + e;
        for (int r = 0; r < 8; ++r) {
          float z = acc[r] + bgv;
          gp[r * CCH] = (_Float16)__builtin_amdgcn_rcpf(1.0f + __expf(-z));
        }
      } else {                                   // q/k/v -> (s,h,l,d), rows stride DH
        int hh = nt >> 1;                        // head, constant per n-tile
        int d  = e & 31;
        float* dp = Dst[w] + (((s * NH + hh) * LSEQ) + l0) * DH + d;
        for (int r = 0; r < 8; ++r) dp[r * DH] = acc[r];
      }
    }
  }
}

// ---------------------------------------------------------------
// Kernel 3: fused inception-depthwise-conv + flash attention.
// grid = 1024 blocks (one per (s,h)) * 256 threads (8 waves).
// ---------------------------------------------------------------
struct ConvW { const float* w[3][3]; const float* b[3][3]; }; // [tensor q/k/v][quadrant-1]

__device__ __forceinline__ float dwconv(const float* __restrict__ base, int l, int d,
                                        int kk, const float* __restrict__ w,
                                        const float* __restrict__ bb)
{
  if (kk == 0) return base[l * DH + d];
  int hw = kk >> 1;
  float acc = bb[d];
  for (int t = 0; t < kk; ++t) {
    int ll = l + t - hw;
    if (0 <= ll && ll < LSEQ) acc += w[d * kk + t] * base[ll * DH + d];
  }
  return acc;
}

__global__ __launch_bounds__(256) void attn_kernel(
    const float* __restrict__ qraw, const float* __restrict__ kraw,
    const float* __restrict__ vraw, const float* __restrict__ bias,
    const float* __restrict__ seq_mask, ConvW cw, _Float16* __restrict__ wa)
{
  __shared__ _Float16 Qs[LSEQ * DH];   // [l][d]  (query, pre-scaled)
  __shared__ _Float16 Kt[DH * LSEQ];   // [l][d]
  __shared__ _Float16 Vs[LSEQ * DH];   // [d][l]
  __shared__ _Float16 Plds[8 * 16 * 32];
  __shared__ float    maskv[LSEQ];

  int bid = blockIdx.x;
  int s = bid >> 2, h = bid & 3;
  int tid = threadIdx.x, wave = tid >> 5, lane = tid & 31;
  int rlo = lane & 15, khalf = lane >> 4;
  int qd = s >> 6;                      // 0:identity 1:k3 2:k5 3:k7
  int kk = (qd == 0) ? 0 : (2 * qd + 1);

  const float* qb = qraw + ((size_t)(s * NH + h) * LSEQ) * DH;
  const float* kb = kraw + ((size_t)(s * NH + h) * LSEQ) * DH;
  const float* vb = vraw + ((size_t)(s * NH + h) * LSEQ) * DH;

  if (tid < LSEQ) maskv[tid] = seq_mask[tid];
  for (int idx = tid; idx < LSEQ * DH; idx += 256) {
    int l = idx >> 5, d = idx & 31;
    float qv = dwconv(qb, l, d, kk, qd ? cw.w[0][qd - 1] : nullptr, qd ? cw.b[0][qd - 1] : nullptr);
    float kv = dwconv(kb, l, d, kk, qd ? cw.w[1][qd - 1] : nullptr, qd ? cw.b[1][qd - 1] : nullptr);
    float vv = dwconv(vb, l, d, kk, qd ? cw.w[2][qd - 1] : nullptr, qd ? cw.b[2][qd - 1] : nullptr);
    Qs[l * DH + d] = (_Float16)(qv * 0.17677669529663689f);  // D^-0.5
    Kt[l * DH + d] = (_Float16)kv;
    Vs[d * LSEQ + l] = (_Float16)vv;
  }
  __syncthreads();

  const float* biasp = bias + (size_t)h * NPIX;
  _Float16* P = Plds + wave * 512;

  for (int t = 0; t < 2; ++t) {                 // 2 query tiles per wave
    int q0 = (wave * 2 + t) * 16;
    __builtin_prefetch(biasp + (q0 + rlo) * LSEQ, 0, 1);  // global_prefetch_b8
    FragU qa;
    const _Float16* qp = Qs + (q0 + rlo) * DH + khalf * 8;
    qa.q[0] = *(const u32x4*)qp;
    qa.q[1] = *(const u32x4*)(qp + 16);

    float m[8], lsum[8];
    v8f o0 = {}, o1 = {};
    for (int r = 0; r < 8; ++r) { m[r] = -1e30f; lsum[r] = 0.0f; }

    for (int kc = 0; kc < 8; ++kc) {            // 32 keys per iteration
      int k0 = kc * 32;
      int dlane = rlo + 16 * khalf;             // unused; QK^T B lane holds key column k0+rlo, d along elements
      const _Float16* kp0 = Kt + (k0 + rlo) * DH + khalf * 8;
      FragU b0, b1;
      b0.q[0] = *(const u32x4*)kp0;             b0.q[1] = *(const u32x4*)(kp0 + 16);
      b1.q[0] = *(const u32x4*)(kp0 + 16 * DH); b1.q[1] = *(const u32x4*)(kp0 + 16 * DH + 16);
      v8f s0 = {}, s1 = {};
      s0 = wmma_f16(qa.h, b0.h, s0);
      s1 = wmma_f16(qa.h, b1.h, s1);

      // hoist V B-frags (depend only on k0) to overlap with softmax VALU work
      int klane = k0 + rlo + 16 * khalf;        // unused; B lane holds d column rlo, keys along elements
      const _Float16* vp = Vs + rlo * LSEQ + k0 + khalf * 8;
      FragU vb0, vb1;
      vb0.q[0] = *(const u32x4*)vp;               vb0.q[1] = *(const u32x4*)(vp + 16);
      vb1.q[0] = *(const u32x4*)(vp + 16 * LSEQ); vb1.q[1] = *(const u32x4*)(vp + 16 * LSEQ + 16);

      int key0 = k0 + rlo, key1 = key0 + 16;
      float mv0 = maskv[key0], mv1 = maskv[key1];
      const float* bp0 = biasp + (q0 + 8 * khalf) * LSEQ + key0;
      float sv0[8], sv1[8], cmax[8], csum[8];
      for (int r = 0; r < 8; ++r) {
        float a0 = s0[r] + bp0[r * LSEQ];
        float a1 = s1[r] + bp0[r * LSEQ + 16];
        sv0[r] = (mv0 > 0.0f) ? a0 : MASKVAL;
        sv1[r] = (mv1 > 0.0f) ? a1 : MASKVAL;
        cmax[r] = fmaxf(sv0[r], sv1[r]);
      }
      for (int r = 0; r < 8; ++r) cmax[r] = redmax16(cmax[r]);
      _Float16* Pp = P + (8 * khalf) * 32 + rlo;
      for (int r = 0; r < 8; ++r) {
        float mn = fmaxf(m[r], cmax[r]);
        float sc = __expf(m[r] - mn);
        float p0 = __expf(sv0[r] - mn);
        float p1 = __expf(sv1[r] - mn);
        m[r] = mn;
        o0[r] *= sc; o1[r] *= sc;
        lsum[r] *= sc;
        csum[r] = p0 + p1;
        Pp[r * 32]      = (_Float16)p0;
        Pp[r * 32 + 16] = (_Float16)p1;
      }
      for (int r = 0; r < 8; ++r) lsum[r] += redsum16(csum[r]);
      asm volatile("s_wait_dscnt 0" ::: "memory");   // intra-wave P RAW guard

      FragU pa;
      const _Float16* pp = P + rlo * 32 + khalf * 8;
      pa.q[0] = *(const u32x4*)pp;
      pa.q[1] = *(const u32x4*)(pp + 16);
      o0 = wmma_f16(pa.h, vb0.h, o0);
      o1 = wmma_f16(pa.h, vb1.h, o1);
    }

    _Float16* wab = wa + (size_t)(s * LSEQ + q0 + 8 * khalf) * CCH + h * DH + rlo;
    for (int r = 0; r < 8; ++r) {
      float rc = __builtin_amdgcn_rcpf(lsum[r] + 1e-20f);
      wab[r * CCH]      = (_Float16)(o0[r] * rc);
      wab[r * CCH + 16] = (_Float16)(o1[r] * rc);
    }
  }
}

// ---------------------------------------------------------------
// Kernel 4: out = (wa * gate) @ wo + bo.  grid 512 * 256 threads.
// wa/gate are f16; A = wa*gate via packed f16 mul, B = wo f16 in LDS.
// ---------------------------------------------------------------
__global__ __launch_bounds__(256) void out_proj_kernel(
    const _Float16* __restrict__ wa, const _Float16* __restrict__ gate,
    const float* __restrict__ wo, const float* __restrict__ bo,
    float* __restrict__ out)
{
  __shared__ _Float16 wlds[CCH * CCH];
  int tid = threadIdx.x, wave = tid >> 5, lane = tid & 31;
  int rlo = lane & 15, khalf = lane >> 4;
  int row0 = blockIdx.x * 128 + wave * 16;

  for (int idx = tid; idx < CCH * CCH; idx += 256) wlds[(idx & (CCH - 1)) * CCH + (idx >> 7)] = (_Float16)wo[idx];
  __syncthreads();

  FragU A[4];
  int row = row0 + rlo;
  for (int kc = 0; kc < 4; ++kc) {
    int base = row * CCH + kc * 32 + khalf * 8;      // A kpat: e<8 -> +e, e>=8 -> +16+e
    v8h wa0 = *(const v8h*)(wa + base);
    v8h wa1 = *(const v8h*)(wa + base + 16);
    v8h g0  = *(const v8h*)(gate + base);
    v8h g1  = *(const v8h*)(gate + base + 16);
    v8h p0 = wa0 * g0;                                // v_pk_mul_f16
    v8h p1 = wa1 * g1;
    for (int e = 0; e < 8; ++e) { A[kc].h[e] = p0[e]; A[kc].h[8 + e] = p1[e]; }
  }
  int koff = rlo + 16 * khalf;
  for (int nt = 0; nt < 8; ++nt) {
    FragU B[4];
    for (int kc = 0; kc < 4; ++kc) {
      const _Float16* bp = wlds + (nt * 16 + rlo) * CCH + kc * 32 + khalf * 8;
      B[kc].q[0] = *(const u32x4*)bp;
      B[kc].q[1] = *(const u32x4*)(bp + 16);
    }
    v8f acc = {};
    for (int kc = 0; kc < 4; ++kc) acc = wmma_f16(A[kc].h, B[kc].h, acc);

    int col = nt * 16 + rlo;
    float bb = bo[col];
    float* op = out + (row0 + 8 * khalf) * CCH + col;
    for (int r = 0; r < 8; ++r) op[r * CCH] = acc[r] + bb;
  }
}

// ---------------------------------------------------------------
extern "C" void kernel_launch(void* const* d_in, const int* in_sizes, int n_in,
                              void* d_out, int out_size, void* d_ws, size_t ws_size,
                              hipStream_t stream) {
  const float* pair_act = (const float*)d_in[0];
  const float* seq_mask = (const float*)d_in[1];
  const float* ln_g = (const float*)d_in[2];
  const float* ln_b = (const float*)d_in[3];
  const float* w_pair = (const float*)d_in[4];
  const float* wq = (const float*)d_in[5];
  const float* wk = (const float*)d_in[6];
  const float* wv = (const float*)d_in[7];
  const float* wg = (const float*)d_in[8];
  const float* bg = (const float*)d_in[9];
  const float* wo = (const float*)d_in[10];
  const float* bo = (const float*)d_in[11];

  ConvW cw;
  for (int t = 0; t < 3; ++t)            // q,k,v
    for (int j = 0; j < 3; ++j) {        // kernels 3,5,7
      cw.w[t][j] = (const float*)d_in[12 + t * 6 + j * 2];
      cw.b[t][j] = (const float*)d_in[12 + t * 6 + j * 2 + 1];
    }

  char* ws = (char*)d_ws;
  const size_t XN_OFF   = 0;                                     // 65536*128 f16
  const size_t BIAS_OFF = XN_OFF   + (size_t)NPIX * CCH * 2;     // 4*65536 f32
  const size_t GATE_OFF = BIAS_OFF + (size_t)NH * NPIX * 4;      // 65536*128 f16
  const size_t Q_OFF    = GATE_OFF + (size_t)NPIX * CCH * 2;     // (s,h,l,d) f32
  const size_t K_OFF    = Q_OFF    + (size_t)NPIX * CCH * 4;
  const size_t V_OFF    = K_OFF    + (size_t)NPIX * CCH * 4;
  const size_t WA_OFF   = V_OFF    + (size_t)NPIX * CCH * 4;     // 65536*128 f16

  _Float16* xn    = (_Float16*)(ws + XN_OFF);
  float* biasb    = (float*)(ws + BIAS_OFF);
  _Float16* gateb = (_Float16*)(ws + GATE_OFF);
  float* qraw     = (float*)(ws + Q_OFF);
  float* kraw     = (float*)(ws + K_OFF);
  float* vraw     = (float*)(ws + V_OFF);
  _Float16* wab   = (_Float16*)(ws + WA_OFF);

  ln_bias_kernel<<<NPIX, 128, 0, stream>>>(pair_act, ln_g, ln_b, w_pair, xn, biasb);
  gemm_qkvg_kernel<<<512, 256, 0, stream>>>(xn, wq, wk, wv, wg, bg, qraw, kraw, vraw, gateb);
  attn_kernel<<<LSEQ * NH, 256, 0, stream>>>(qraw, kraw, vraw, biasb, seq_mask, cw, wab);
  out_proj_kernel<<<512, 256, 0, stream>>>(wab, gateb, wo, bo, (float*)d_out);
}
